// GCN_9887014715723
// MI455X (gfx1250) — matrix operand order repaired and checked by reading, not compiled, not run on hardware
//
#include <hip/hip_runtime.h>
#include <hip/hip_bf16.h>
#include <math.h>

typedef __attribute__((ext_vector_type(16))) _Float16 v16h;
typedef __attribute__((ext_vector_type(8)))  float    v8f;

#define IN_F  512
#define H1_F  16
#define OUT_F 40

// ---------------- degree / norm ----------------
__global__ void k_deg_init(float* __restrict__ deg, int n) {
    int i = blockIdx.x * blockDim.x + threadIdx.x;
    if (i < n) deg[i] = 1.0f;   // self-loop contributes 1
}

__global__ void k_deg_count(const int* __restrict__ dst, float* __restrict__ deg, int E) {
    int e = blockIdx.x * blockDim.x + threadIdx.x;
    if (e < E) atomicAdd(&deg[dst[e]], 1.0f);
}

__global__ void k_deg_rsqrt(float* __restrict__ deg, int n) {
    int i = blockIdx.x * blockDim.x + threadIdx.x;
    if (i < n) deg[i] = rsqrtf(deg[i]);   // deg >= 1 always (self-loops)
}

// ---------------- GEMM1: h1 = x @ W1  (K=512, N=16), split-f16 WMMA ----------------
__global__ __launch_bounds__(256) void k_gemm1(const float* __restrict__ x,
                                               const float* __restrict__ W1,
                                               float* __restrict__ h1, int n) {
    int wave = threadIdx.x >> 5;
    int lane = threadIdx.x & 31;
    int tile = blockIdx.x * 8 + wave;     // one 16-row tile per wave
    int nTiles = n >> 4;                  // n is a multiple of 16
    if (tile >= nTiles) return;           // wave-uniform: EXEC all-ones at WMMA

    int m  = lane & 15;                   // A row / B col / C col
    int hh = lane >> 4;                   // lane half

    const float* arow = x + (size_t)(tile * 16 + m) * IN_F;
    v8f c = {};
    for (int k0 = 0; k0 < IN_F; k0 += 32) {
        v16h ah, al, bh, bl;
        #pragma unroll
        for (int e = 0; e < 16; ++e) {
            // A(16x32) 16-bit layout: K = ((e&8)<<1) + 8*half + (e&7)
            int ka = ((e & 8) << 1) + (hh << 3) + (e & 7);
            float av = arow[k0 + ka];
            _Float16 ahi = (_Float16)av;
            ah[e] = ahi;
            al[e] = (_Float16)(av - (float)ahi);
            // B(32x16) 16-bit layout: K = ((e&8)<<1) + 8*half + (e&7)
            int kb = ((e & 8) << 1) + (hh << 3) + (e & 7);
            float bv = W1[(size_t)(k0 + kb) * H1_F + m];
            _Float16 bhi = (_Float16)bv;
            bh[e] = bhi;
            bl[e] = (_Float16)(bv - (float)bhi);
        }
        c = __builtin_amdgcn_wmma_f32_16x16x32_f16(false, ah, false, bh, (short)0, c, false, false);
        c = __builtin_amdgcn_wmma_f32_16x16x32_f16(false, ah, false, bl, (short)0, c, false, false);
        c = __builtin_amdgcn_wmma_f32_16x16x32_f16(false, al, false, bh, (short)0, c, false, false);
    }
    for (int r = 0; r < 8; ++r) {
        int mr = r + 8 * hh;              // C/D row layout
        h1[(size_t)(tile * 16 + mr) * H1_F + m] = c[r];
    }
}

// ---------------- self-loop init: agg = h * dinv^2 ----------------
__global__ void k_selfloop(const float* __restrict__ h, const float* __restrict__ dinv,
                           float* __restrict__ agg, int total, int feat) {
    int t = blockIdx.x * blockDim.x + threadIdx.x;
    if (t >= total) return;
    int i = t / feat;
    float w = dinv[i] * dinv[i];
    agg[t] = h[t] * w;
}

// ---------------- scatter layer 1: 16 features per edge ----------------
__global__ void k_scatter16(const int* __restrict__ src, const int* __restrict__ dst,
                            const float* __restrict__ dinv, const float* __restrict__ h1,
                            float* __restrict__ agg, int E) {
    int e = blockIdx.x * blockDim.x + threadIdx.x;
    if (e >= E) return;
    int s = src[e], d = dst[e];
    float w = dinv[s] * dinv[d];
    const float4* hs = (const float4*)(h1 + (size_t)s * H1_F);
    float* out = agg + (size_t)d * H1_F;
    #pragma unroll
    for (int q = 0; q < 4; ++q) {
        float4 v = hs[q];
        atomicAdd(out + q * 4 + 0, v.x * w);
        atomicAdd(out + q * 4 + 1, v.y * w);
        atomicAdd(out + q * 4 + 2, v.z * w);
        atomicAdd(out + q * 4 + 3, v.w * w);
    }
}

// ---------------- bias + relu in place ----------------
__global__ void k_bias_relu16(float* __restrict__ agg, const float* __restrict__ b1, int total) {
    int t = blockIdx.x * blockDim.x + threadIdx.x;
    if (t >= total) return;
    float v = agg[t] + b1[t & 15];
    agg[t] = v > 0.0f ? v : 0.0f;
}

// ---------------- GEMM2: h2 = xr @ W2  (K=16, N=40 -> 3 col tiles) ----------------
__global__ __launch_bounds__(256) void k_gemm2(const float* __restrict__ xr,
                                               const float* __restrict__ W2,
                                               float* __restrict__ h2, int n) {
    int wave = threadIdx.x >> 5;
    int lane = threadIdx.x & 31;
    int g = blockIdx.x * 8 + wave;
    int nRowTiles = n >> 4;
    if (g >= nRowTiles * 3) return;       // wave-uniform
    int rowTile = g / 3;
    int colTile = g - rowTile * 3;

    int m   = lane & 15;
    int hh  = lane >> 4;
    int col = colTile * 16 + m;

    const float* arow = xr + (size_t)(rowTile * 16 + m) * H1_F;
    v16h ah = {}, al = {}, bh = {}, bl = {};
    #pragma unroll
    for (int e = 0; e < 8; ++e) {
        // elements e<8 map to K = 8*half + e (< 16); e>=8 map to K>=16 -> stay zero
        float av = arow[(hh << 3) + e];
        _Float16 ahi = (_Float16)av;
        ah[e] = ahi;
        al[e] = (_Float16)(av - (float)ahi);
    }
    {
        // B elements e<8 map to K = 8*half + e (< 16); e>=8 map to K>=16 -> stay zero
        #pragma unroll
        for (int e = 0; e < 8; ++e) {
            float bv = (col < OUT_F) ? W2[(size_t)((hh << 3) + e) * OUT_F + col] : 0.0f;
            _Float16 bhi = (_Float16)bv;
            bh[e] = bhi;
            bl[e] = (_Float16)(bv - (float)bhi);
        }
    }
    v8f c = {};
    c = __builtin_amdgcn_wmma_f32_16x16x32_f16(false, ah, false, bh, (short)0, c, false, false);
    c = __builtin_amdgcn_wmma_f32_16x16x32_f16(false, ah, false, bl, (short)0, c, false, false);
    c = __builtin_amdgcn_wmma_f32_16x16x32_f16(false, al, false, bh, (short)0, c, false, false);

    if (col < OUT_F) {
        for (int r = 0; r < 8; ++r) {
            int mr = r + 8 * hh;
            h2[(size_t)(rowTile * 16 + mr) * OUT_F + col] = c[r];
        }
    }
}

// ---------------- scatter layer 2: 40 features, 4 threads/edge x 10 cols ----------------
__global__ void k_scatter40(const int* __restrict__ src, const int* __restrict__ dst,
                            const float* __restrict__ dinv, const float* __restrict__ h2,
                            float* __restrict__ agg, int E) {
    int t = blockIdx.x * blockDim.x + threadIdx.x;
    int e = t >> 2;
    if (e >= E) return;
    int part = t & 3;
    int s = src[e], d = dst[e];
    float w = dinv[s] * dinv[d];
    const float* hs = h2 + (size_t)s * OUT_F + part * 10;
    float* out = agg + (size_t)d * OUT_F + part * 10;
    #pragma unroll
    for (int c = 0; c < 10; ++c) atomicAdd(out + c, hs[c] * w);
}

// ---------------- bias + log_softmax (one row per thread) ----------------
__global__ void k_bias_lsm(const float* __restrict__ agg, const float* __restrict__ b2,
                           float* __restrict__ out, int n) {
    int i = blockIdx.x * blockDim.x + threadIdx.x;
    if (i >= n) return;
    const float* row = agg + (size_t)i * OUT_F;
    float v[OUT_F];
    float mx = -INFINITY;
    #pragma unroll
    for (int c = 0; c < OUT_F; ++c) { v[c] = row[c] + b2[c]; mx = fmaxf(mx, v[c]); }
    float s = 0.0f;
    #pragma unroll
    for (int c = 0; c < OUT_F; ++c) s += expf(v[c] - mx);
    float lse = mx + logf(s);
    float* orow = out + (size_t)i * OUT_F;
    #pragma unroll
    for (int c = 0; c < OUT_F; ++c) orow[c] = v[c] - lse;
}

extern "C" void kernel_launch(void* const* d_in, const int* in_sizes, int n_in,
                              void* d_out, int out_size, void* d_ws, size_t ws_size,
                              hipStream_t stream) {
    const float* x  = (const float*)d_in[0];
    const int*   ei = (const int*)d_in[1];
    const float* W1 = (const float*)d_in[2];
    const float* b1 = (const float*)d_in[3];
    const float* W2 = (const float*)d_in[4];
    const float* b2 = (const float*)d_in[5];
    float* out = (float*)d_out;

    int n = in_sizes[0] / IN_F;   // 100000
    int E = in_sizes[1] / 2;      // 3200000
    const int* src = ei;
    const int* dst = ei + E;

    // workspace layout (floats): dinv[n] | h1[n*16] | agg1[n*16] | h2[n*40] | agg2[n*40]
    float* ws   = (float*)d_ws;
    float* dinv = ws;
    float* h1   = dinv + n;
    float* agg1 = h1   + (size_t)n * H1_F;
    float* h2   = agg1 + (size_t)n * H1_F;
    float* agg2 = h2   + (size_t)n * OUT_F;

    const int B = 256;
    int nTiles = n / 16;

    // normalization coefficients
    k_deg_init <<<(n + B - 1) / B, B, 0, stream>>>(dinv, n);
    k_deg_count<<<(E + B - 1) / B, B, 0, stream>>>(dst, dinv, E);
    k_deg_rsqrt<<<(n + B - 1) / B, B, 0, stream>>>(dinv, n);

    // layer 1
    k_gemm1<<<(nTiles + 7) / 8, B, 0, stream>>>(x, W1, h1, n);
    k_selfloop<<<((n * H1_F) + B - 1) / B, B, 0, stream>>>(h1, dinv, agg1, n * H1_F, H1_F);
    k_scatter16<<<(E + B - 1) / B, B, 0, stream>>>(src, dst, dinv, h1, agg1, E);
    k_bias_relu16<<<((n * H1_F) + B - 1) / B, B, 0, stream>>>(agg1, b1, n * H1_F);

    // layer 2
    k_gemm2<<<(nTiles * 3 + 7) / 8, B, 0, stream>>>(agg1, W2, h2, n);
    k_selfloop<<<((n * OUT_F) + B - 1) / B, B, 0, stream>>>(h2, dinv, agg2, n * OUT_F, OUT_F);
    k_scatter40<<<((E * 4) + B - 1) / B, B, 0, stream>>>(src, dst, dinv, h2, agg2, E);

    // bias + log_softmax
    k_bias_lsm<<<(n + B - 1) / B, B, 0, stream>>>(agg2, b2, out, n);
}
